// Multi_Adaptive_Hypergraph_65008624993104
// MI455X (gfx1250) — compile-verified, matrix operand order checked by reading, not run on hardware
//
#include <hip/hip_runtime.h>
#include <hip/hip_bf16.h>

// ---------------------------------------------------------------------------
// Multi_Adaptive_Hypergraph on MI455X (gfx1250, wave32)
//   Pass1: WMMA f16 GEMM (node @ edge^T) -> LDS row block -> softmax(relu(3a))
//          -> top-16 threshold -> EMA blend -> unnormalized adj + global min/max
//   Pass2: global minmax normalize, binarize, fused upsample scatter, retain OR
//          (128-bit vectorized, non-temporal: outputs are write-once, keep L2 clean)
//   Pass3: retain vectors
// ---------------------------------------------------------------------------

typedef _Float16 v16h __attribute__((ext_vector_type(16)));
typedef float    v8f  __attribute__((ext_vector_type(8)));
typedef float    v4f  __attribute__((ext_vector_type(4)));

#define SEQ_LEN 32768
#define FUSED_W 3584

// output layout (floats), reference return order
constexpr size_t SZ_FUSED = (size_t)SEQ_LEN * FUSED_W;      // 117,440,512
constexpr size_t OFS_FUSED_H = 0;
constexpr size_t OFS_H0   = OFS_FUSED_H + SZ_FUSED;
constexpr size_t OFS_H1   = OFS_H0 + 32768ull * 2048;
constexpr size_t OFS_H2   = OFS_H1 + 16384ull * 1024;
constexpr size_t OFS_BIN0 = OFS_H2 + 8192ull * 512;
constexpr size_t OFS_BIN1 = OFS_BIN0 + 32768ull * 2048;
constexpr size_t OFS_BIN2 = OFS_BIN1 + 16384ull * 1024;
constexpr size_t OFS_FBIN = OFS_BIN2 + 8192ull * 512;
constexpr size_t OFS_R0   = OFS_FBIN + SZ_FUSED;
constexpr size_t OFS_R1   = OFS_R0 + 2048;
constexpr size_t OFS_R2   = OFS_R1 + 1024;
constexpr size_t OFS_FR   = OFS_R2 + 512;

// workspace (uints): [2*i]=min bits, [2*i+1]=max bits for level i; [6..6+3584) retain flags
#define WS_RETAIN 6
#define WS_TOTAL  (WS_RETAIN + FUSED_W)

static __device__ __forceinline__ float wred_max(float v) {
  #pragma unroll
  for (int o = 16; o > 0; o >>= 1) v = fmaxf(v, __shfl_xor(v, o, 32));
  return v;
}
static __device__ __forceinline__ float wred_min(float v) {
  #pragma unroll
  for (int o = 16; o > 0; o >>= 1) v = fminf(v, __shfl_xor(v, o, 32));
  return v;
}
static __device__ __forceinline__ float wred_sum(float v) {
  #pragma unroll
  for (int o = 16; o > 0; o >>= 1) v += __shfl_xor(v, o, 32);
  return v;
}

static __device__ __forceinline__ v8f wmma16x16x32(v16h a, v16h b, v8f c) {
  // D = A(16x32 f16) * B(32x16 f16) + C(16x16 f32)   [codegen-confirmed builtin]
  return __builtin_amdgcn_wmma_f32_16x16x32_f16(false, a, false, b, (short)0, c,
                                                false, false);
}

__global__ void init_ws_kernel(unsigned int* ws) {
  int i = blockIdx.x * blockDim.x + threadIdx.x;
  if (i >= WS_TOTAL) return;
  if (i < WS_RETAIN) ws[i] = (i & 1) ? 0u : 0x7F7FFFFFu;  // max:=0, min:=FLT_MAX
  else               ws[i] = 0u;
}

// One block of 256 threads (8 waves) per 16-row tile.
template <int H>
__global__ __launch_bounds__(256) void pass1_kernel(const float* __restrict__ node,
                                                    const float* __restrict__ edge,
                                                    const float* __restrict__ adj_init,
                                                    float* __restrict__ outH,
                                                    unsigned int* __restrict__ mm) {
  __shared__ float    a_lds[16 * H];        // full 16-row block of `a` (f32)
  __shared__ _Float16 node_lds[16 * 64];    // node tile, f16

  const int row_base = blockIdx.x * 16;

  // Stage node tile (16 x 64) as f16
  for (int i = threadIdx.x; i < 16 * 64; i += 256)
    node_lds[i] = (_Float16)node[(size_t)(row_base + (i >> 6)) * 64 + (i & 63)];
  __syncthreads();

  const int lane = threadIdx.x & 31;
  const int wid  = threadIdx.x >> 5;
  const int hi   = lane >> 4;       // lane half selects K/M sub-block
  const int kc   = hi * 8;
  const int nn   = lane & 15;

  // A operands (16x32 f16, K blocks 0..31 and 32..63) — shared by all tiles
  v16h a0, a1;
  #pragma unroll
  for (int j = 0; j < 16; ++j) {
    const int k = ((j < 8) ? j : j + 8) + kc;   // ISA 16-bit A layout
    a0[j] = node_lds[(lane & 15) * 64 + k];
    a1[j] = node_lds[(lane & 15) * 64 + 32 + k];
  }

  // GEMM: each wave handles N-tiles nt = wid, wid+8, ...
  const int ntiles = H / 16;
  for (int nt = wid; nt < ntiles; nt += 8) {
    if (nt + 8 < ntiles)
      __builtin_prefetch(edge + (size_t)((nt + 8) * 16 + nn) * 64, 0, 0);
    const float* erow = edge + (size_t)(nt * 16 + nn) * 64;  // B[k][n] = edge[n][k]
    v16h b0, b1;
    #pragma unroll
    for (int j = 0; j < 16; ++j) {
      const int k = ((j < 8) ? j : j + 8) + kc;
      b0[j] = (_Float16)erow[k];
      b1[j] = (_Float16)erow[32 + k];
    }
    v8f acc = {};
    acc = wmma16x16x32(a0, b0, acc);
    acc = wmma16x16x32(a1, b1, acc);
    // C/D layout: lane<16 -> M=j, lane>=16 -> M=j+8; N = lane&15
    #pragma unroll
    for (int j = 0; j < 8; ++j)
      a_lds[(j + hi * 8) * H + nt * 16 + nn] = acc[j];
  }
  __syncthreads();

  // Row-wise softmax(relu(3a)) + top-16 threshold + blend. Wave w owns rows 2w, 2w+1.
  for (int rr = 0; rr < 2; ++rr) {
    const int r = wid * 2 + rr;
    const float* arow = a_lds + r * H;

    // max of relu(3a)  (>= 0)
    float mx = 0.f;
    for (int e = lane; e < H; e += 32) {
      float v = 3.f * arow[e];
      mx = fmaxf(mx, v > 0.f ? v : 0.f);
    }
    mx = wred_max(mx);

    // sum exp
    float s = 0.f;
    for (int e = lane; e < H; e += 32) {
      float v = 3.f * arow[e];
      if (v < 0.f) v = 0.f;
      s += __expf(v - mx);
    }
    s = wred_sum(s);
    const float inv = 1.f / s;

    // per-lane sorted top-16 of relu(3a) (softmax is monotone => same top-k set)
    float arr[16];
    #pragma unroll
    for (int j = 0; j < 16; ++j) arr[j] = -3.4e38f;
    for (int e = lane; e < H; e += 32) {
      float v = 3.f * arow[e];
      if (v < 0.f) v = 0.f;
      if (v > arr[15]) {
        int p = 15;
        while (p > 0 && v > arr[p - 1]) { arr[p] = arr[p - 1]; --p; }
        arr[p] = v;
      }
    }
    // merge across 32 lanes: pop global max 16 times -> 16th largest = threshold
    int ptr = 0;
    float th = 0.f;
    for (int it = 0; it < 16; ++it) {
      float cand = (ptr < 16) ? arr[ptr] : -3.4e38f;
      float cm = wred_max(cand);
      unsigned long long bal = __ballot(cand == cm);
      if (lane == (__ffsll(bal) - 1)) ++ptr;  // exactly one popper (tie -> lowest lane)
      th = cm;
    }

    // masked softmax, EMA blend, write unnormalized adj, track global min/max.
    // adj_init is read exactly once on-device -> non-temporal load.
    float lmin = 3.4e38f, lmax = 0.f;
    const size_t rowofs = (size_t)(row_base + r) * H;
    for (int e = lane; e < H; e += 32) {
      float v = 3.f * arow[e];
      if (v < 0.f) v = 0.f;
      float cur = (v >= th) ? __expf(v - mx) * inv : 0.f;
      float ai = __builtin_nontemporal_load(adj_init + rowofs + e);
      float adj = 0.5f * cur + 0.5f * ai;
      outH[rowofs + e] = adj;  // reread by pass2: keep regular temporal
      lmin = fminf(lmin, adj);
      lmax = fmaxf(lmax, adj);
    }
    lmin = wred_min(lmin);
    lmax = wred_max(lmax);
    if (lane == 0) {  // adj >= 0, so float bits compare like unsigned
      atomicMin(&mm[0], __float_as_uint(lmin));
      atomicMax(&mm[1], __float_as_uint(lmax));
    }
  }
}

// Bandwidth-critical pass: 4 consecutive columns per thread (b128 loads/stores),
// non-temporal stores so the ~1.6 GB of write-once output bypasses L2 retention.
template <int N, int H, int FACTOR, int COLOFS>
__global__ __launch_bounds__(256) void pass2_kernel(float* __restrict__ outH,
                                                    float* __restrict__ outBin,
                                                    float* __restrict__ fusedH,
                                                    float* __restrict__ fusedBin,
                                                    const unsigned int* __restrict__ mm,
                                                    unsigned int* __restrict__ retain) {
  const size_t idx = ((size_t)blockIdx.x * 256 + threadIdx.x) * 4;
  if (idx >= (size_t)N * H) return;
  const int n = (int)(idx / H);
  const int e = (int)(idx % H);            // H % 4 == 0 -> all 4 elems in one row
  const float mn   = __uint_as_float(mm[0]);
  const float mxv  = __uint_as_float(mm[1]);
  const float rinv = 1.f / (mxv - mn);

  v4f v = __builtin_nontemporal_load((const v4f*)(outH + idx));  // last use
  v4f norm = (v - mn) * rinv;
  v4f b;
  #pragma unroll
  for (int c = 0; c < 4; ++c) b[c] = (norm[c] > 0.5f) ? 1.f : 0.f;

  __builtin_nontemporal_store(norm, (v4f*)(outH + idx));
  __builtin_nontemporal_store(b, (v4f*)(outBin + idx));
  #pragma unroll
  for (int f = 0; f < FACTOR; ++f) {
    const size_t fi = ((size_t)n * FACTOR + f) * FUSED_W + COLOFS + e;
    __builtin_nontemporal_store(norm, (v4f*)(fusedH + fi));
    __builtin_nontemporal_store(b, (v4f*)(fusedBin + fi));
  }
  #pragma unroll
  for (int c = 0; c < 4; ++c)
    if (b[c] != 0.f) atomicOr(&retain[COLOFS + e + c], 1u);  // rare (top-k only)
}

__global__ void retain_kernel(const unsigned int* __restrict__ flags,
                              float* __restrict__ out) {
  const int e = blockIdx.x * blockDim.x + threadIdx.x;
  if (e >= FUSED_W) return;
  const float fv = flags[e] ? 1.f : 0.f;
  out[OFS_FR + e] = fv;
  if (e < 2048)       out[OFS_R0 + e] = fv;
  else if (e < 3072)  out[OFS_R1 + (e - 2048)] = fv;
  else                out[OFS_R2 + (e - 3072)] = fv;
}

extern "C" void kernel_launch(void* const* d_in, const int* in_sizes, int n_in,
                              void* d_out, int out_size, void* d_ws, size_t ws_size,
                              hipStream_t stream) {
  const float* node0 = (const float*)d_in[0];
  const float* node1 = (const float*)d_in[1];
  const float* node2 = (const float*)d_in[2];
  const float* edge0 = (const float*)d_in[3];
  const float* edge1 = (const float*)d_in[4];
  const float* edge2 = (const float*)d_in[5];
  const float* adj0  = (const float*)d_in[6];
  const float* adj1  = (const float*)d_in[7];
  const float* adj2  = (const float*)d_in[8];

  float* out = (float*)d_out;
  unsigned int* ws = (unsigned int*)d_ws;
  unsigned int* retain = ws + WS_RETAIN;

  init_ws_kernel<<<(WS_TOTAL + 255) / 256, 256, 0, stream>>>(ws);

  // Pass 1: WMMA GEMM + softmax/top-k/blend + global min/max
  pass1_kernel<2048><<<32768 / 16, 256, 0, stream>>>(node0, edge0, adj0, out + OFS_H0, ws + 0);
  pass1_kernel<1024><<<16384 / 16, 256, 0, stream>>>(node1, edge1, adj1, out + OFS_H1, ws + 2);
  pass1_kernel<512><<<8192 / 16, 256, 0, stream>>>(node2, edge2, adj2, out + OFS_H2, ws + 4);

  // Pass 2: normalize, binarize, fused scatter, retain flags (vec4, non-temporal)
  pass2_kernel<32768, 2048, 1, 0>
      <<<(32768 * 2048) / (256 * 4), 256, 0, stream>>>(out + OFS_H0, out + OFS_BIN0,
                                                       out + OFS_FUSED_H, out + OFS_FBIN,
                                                       ws + 0, retain);
  pass2_kernel<16384, 1024, 2, 2048>
      <<<(16384 * 1024) / (256 * 4), 256, 0, stream>>>(out + OFS_H1, out + OFS_BIN1,
                                                       out + OFS_FUSED_H, out + OFS_FBIN,
                                                       ws + 2, retain);
  pass2_kernel<8192, 512, 4, 3072>
      <<<(8192 * 512) / (256 * 4), 256, 0, stream>>>(out + OFS_H2, out + OFS_BIN2,
                                                     out + OFS_FUSED_H, out + OFS_FBIN,
                                                     ws + 4, retain);

  // Pass 3: retain vectors
  retain_kernel<<<(FUSED_W + 255) / 256, 256, 0, stream>>>(retain, out);
}
